// MultiHeadAttention_22883585753377
// MI455X (gfx1250) — matrix operand order repaired and checked by reading, not compiled, not run on hardware
//
#include <hip/hip_runtime.h>
#include <hip/hip_bf16.h>
#include <stdint.h>

// ---------------------------------------------------------------------------
// MHA with softmax-over-heads, MI455X (gfx1250), bf16 WMMA pipeline with
// async global->LDS staging (ASYNCcnt) and software-pipelined fragments.
//   B=2, S=2048, D=1024, H=16, head_dim=64, scale = 1/sqrt(1024) = 1/32.
// ---------------------------------------------------------------------------

typedef __bf16 bf16_t;
typedef __attribute__((ext_vector_type(16))) __bf16 v16bf;
typedef __attribute__((ext_vector_type(8)))  __bf16 v8bf;
typedef __attribute__((ext_vector_type(8)))  float  v8f;

#define WMMA_BF16(a, b, c) \
  __builtin_amdgcn_wmma_f32_16x16x32_bf16(false, (a), false, (b), (short)0, (c), false, false)

static constexpr int kB  = 2;
static constexpr int kS  = 2048;
static constexpr int kD  = 1024;
static constexpr int kH  = 16;
static constexpr int kDH = 64;
static constexpr int kN  = 1024;     // projected dim for q/k/v/out
static constexpr int kM  = kB * kS;  // 4096 rows

// ---- CDNA5 async global->LDS (ASYNCcnt) helpers; inline asm is portable
// ---- across ROCm7.2 / amdgpu-toolchain (bypasses the clang TDM builtins).
static __device__ __forceinline__ void async_load_b128(unsigned lds_off,
                                                       const void* gaddr) {
  asm volatile("global_load_async_to_lds_b128 %0, %1, off"
               :: "v"(lds_off), "v"(gaddr) : "memory");
}
static __device__ __forceinline__ void wait_async0() {
  asm volatile("s_wait_asynccnt 0" ::: "memory");
}
static __device__ __forceinline__ unsigned lds_offset(const void* p) {
  // flat LDS aperture: addr[31:0] is the byte offset within the LDS alloc
  return (unsigned)(uintptr_t)p;
}

// ---------------------------------------------------------------------------
// Kernel 1: W [K,N] fp32 -> Wt [N,K] bf16  (transposed so WMMA B-fragments
// are contiguous 32-byte per-lane loads).
// ---------------------------------------------------------------------------
__global__ void wt_transpose_kernel(const float* __restrict__ w,
                                    bf16_t* __restrict__ wt) {
  int idx = blockIdx.x * blockDim.x + threadIdx.x;
  if (idx >= kD * kN) return;
  int k = idx / kN;
  int n = idx % kN;                       // coalesced read along n
  wt[(size_t)n * kD + k] = (bf16_t)w[(size_t)k * kN + n];
}

// ---------------------------------------------------------------------------
// Kernels 2/4: WMMA GEMM with per-wave double-buffered async B staging.
// Block = 4 waves, each wave: 16(M) x 64(N) tile, 4 f32 accumulators.
// Block tile: 16 x 256. Wave-private LDS buffers -> no barriers needed;
// s_wait_asynccnt 0 at loop top covers only the batch issued last iteration,
// so global->LDS traffic overlaps the WMMAs.
// ---------------------------------------------------------------------------

// Stage next B tile: 64 rows x 32 K bf16 (4KB) = 8 async b128 (512B each).
// Lane l covers row (i*8 + l/4), 16-byte chunk (l%4) of the 64B row.
template <typename LoadA, typename Epilogue>
static __device__ __forceinline__ void gemm_core(
    const bf16_t* __restrict__ Wt, bf16_t* btile /* [2][64*32] wave-private */,
    int n0, int lane, LoadA load_a, Epilogue epi) {
  const int lr = lane & 15, hg = lane >> 4;
  const int srow = lane >> 2;        // 0..7 row-in-chunk
  const int scol = (lane & 3) * 8;   // element offset (16B chunks)

  const bf16_t* gbase[8];
  unsigned loff[2][8];
#pragma unroll
  for (int i = 0; i < 8; ++i) {
    gbase[i] = Wt + (size_t)(n0 + i * 8 + srow) * kD + scol;
    loff[0][i] = lds_offset(btile + 0 * 64 * 32 + (i * 8 + srow) * 32 + scol);
    loff[1][i] = lds_offset(btile + 1 * 64 * 32 + (i * 8 + srow) * 32 + scol);
  }

  v8f acc[4] = {v8f{}, v8f{}, v8f{}, v8f{}};

  // prologue: issue tile 0, preload A frag 0
#pragma unroll
  for (int i = 0; i < 8; ++i) async_load_b128(loff[0][i], gbase[i]);
  v16bf a = load_a(0);

  for (int kb = 0; kb < kD; kb += 32) {
    const int buf = (kb >> 5) & 1;
    wait_async0();  // only waits the batch issued one iteration ago
    const bf16_t* bt = btile + buf * 64 * 32;
    v16bf b0 = *(const v16bf*)(bt + (0 * 16 + lr) * 32 + hg * 16);
    v16bf b1 = *(const v16bf*)(bt + (1 * 16 + lr) * 32 + hg * 16);
    v16bf b2 = *(const v16bf*)(bt + (2 * 16 + lr) * 32 + hg * 16);
    v16bf b3 = *(const v16bf*)(bt + (3 * 16 + lr) * 32 + hg * 16);
    v16bf an = a;
    if (kb + 32 < kD) {
      an = load_a(kb + 32);                    // pipeline A one step ahead
#pragma unroll
      for (int i = 0; i < 8; ++i)              // prefetch next B tile -> LDS
        async_load_b128(loff[buf ^ 1][i], gbase[i] + kb + 32);
    }
    acc[0] = WMMA_BF16(a, b0, acc[0]);
    acc[1] = WMMA_BF16(a, b1, acc[1]);
    acc[2] = WMMA_BF16(a, b2, acc[2]);
    acc[3] = WMMA_BF16(a, b3, acc[3]);
    a = an;
  }
  epi(acc);
}

// Projection GEMM: X fp32 [M,K] @ Wt bf16 [N,K] + bias -> bf16 head-split.
//   addr = (b*H + h)*S*64 + s*ss + d*sd   (Q/K: ss=64,sd=1; V: ss=1,sd=S)
__global__ __launch_bounds__(128) void gemm_proj_kernel(
    const float* __restrict__ X, const bf16_t* __restrict__ Wt,
    const float* __restrict__ bias, bf16_t* __restrict__ out,
    int ss, int sd) {
  __shared__ __align__(64) bf16_t btile[4][2][64 * 32];  // 32KB
  const int wave = threadIdx.x >> 5;
  const int lane = threadIdx.x & 31;
  const int lr = lane & 15, hg = lane >> 4;
  const int m0 = blockIdx.x * 16;
  const int n0 = blockIdx.y * 256 + wave * 64;

  const float* arow = X + (size_t)(m0 + lr) * kD;
  auto load_a = [&](int kb) {
    const float* p1 = arow + kb + hg * 16;  // contiguous A layout: lanes16-31 get K+16
    const float* p2 = p1 + 8;
    __builtin_prefetch(arow + kb + 64, 0, 1);
    v16bf a;
#pragma unroll
    for (int i = 0; i < 8; ++i) {
      a[i]     = (bf16_t)p1[i];
      a[8 + i] = (bf16_t)p2[i];
    }
    return a;
  };
  auto epi = [&](v8f* acc) {
#pragma unroll
    for (int j = 0; j < 4; ++j) {
      const int col = n0 + j * 16 + lr;
      const float bb = bias[col];
      const int h = col >> 6, d = col & 63;
#pragma unroll
      for (int r = 0; r < 8; ++r) {
        const int row  = m0 + r + hg * 8;
        const int bidx = row >> 11, srow2 = row & 2047;
        const size_t base = (size_t)(bidx * kH + h) * (size_t)(kS * kDH);
        out[base + (size_t)srow2 * ss + (size_t)d * sd] = (bf16_t)(acc[j][r] + bb);
      }
    }
  };
  gemm_core(Wt, &btile[wave][0][0], n0, lane, load_a, epi);
}

// Output GEMM: Obf bf16 [M,K] @ WoT bf16 [N,K] + bo -> fp32 dense.
__global__ __launch_bounds__(128) void gemm_out_kernel(
    const bf16_t* __restrict__ A, const bf16_t* __restrict__ Wt,
    const float* __restrict__ bias, float* __restrict__ out) {
  __shared__ __align__(64) bf16_t btile[4][2][64 * 32];  // 32KB
  const int wave = threadIdx.x >> 5;
  const int lane = threadIdx.x & 31;
  const int lr = lane & 15, hg = lane >> 4;
  const int m0 = blockIdx.x * 16;
  const int n0 = blockIdx.y * 256 + wave * 64;

  const bf16_t* arow = A + (size_t)(m0 + lr) * kD;
  auto load_a = [&](int kb) {
    __builtin_prefetch(arow + kb + 64, 0, 1);
    v8bf lo = *(const v8bf*)(arow + kb + hg * 16);
    v8bf hi = *(const v8bf*)(arow + kb + 8 + hg * 16);
    v16bf a;
#pragma unroll
    for (int i = 0; i < 8; ++i) { a[i] = lo[i]; a[8 + i] = hi[i]; }
    return a;
  };
  auto epi = [&](v8f* acc) {
#pragma unroll
    for (int j = 0; j < 4; ++j) {
      const int col = n0 + j * 16 + lr;
      const float bb = bias[col];
#pragma unroll
      for (int r = 0; r < 8; ++r) {
        const int row = m0 + r + hg * 8;
        out[(size_t)row * kN + col] = acc[j][r] + bb;
      }
    }
  };
  gemm_core(Wt, &btile[wave][0][0], n0, lane, load_a, epi);
}

// ---------------------------------------------------------------------------
// Kernel 3: attention with softmax over the HEADS axis, streamed over k-tiles.
// Block = 16 waves (one head per wave), 16 query rows, k-tiles of 32 tokens.
// Per k-tile: 4 score WMMA -> LDS -> cross-head max/sum -> 4 output WMMA.
// V-fragments are loaded before the barrier phase so their latency overlaps
// the cross-head reduction.
// ---------------------------------------------------------------------------
__global__ __launch_bounds__(512) void attn_kernel(
    const bf16_t* __restrict__ Qb,   // [B,H,S,64]
    const bf16_t* __restrict__ Kb,   // [B,H,S,64]
    const bf16_t* __restrict__ Vt,   // [B,H,64,S]
    bf16_t* __restrict__ Obf) {      // [B*S, 1024] merged heads
  __shared__ float sc[kH][16][32];   // raw scaled scores, all heads (32KB)
  __shared__ float mred[16][32];     // per-(q,k) max over heads
  __shared__ float rinv[16][32];     // per-(q,k) 1/sum(exp)

  const int h    = threadIdx.x >> 5;   // wave index == head
  const int lane = threadIdx.x & 31;
  const int lr   = lane & 15;
  const int hg   = lane >> 4;
  const int b    = blockIdx.y;
  const int q0   = blockIdx.x * 16;
  const float scale = 0.03125f;        // 1/sqrt(KEY_DIM=1024)

  const size_t head_base = (size_t)(b * kH + h) * (size_t)(kS * kDH);

  // Preload the two q A-fragments (K = 0..31 and 32..63), loop-invariant.
  v16bf aq0, aq1;
  {
    const bf16_t* qrow = Qb + head_base + (size_t)(q0 + lr) * kDH;
    const int klane = hg * 16;
    v8bf x0 = *(const v8bf*)(qrow + 0  + klane);
    v8bf x1 = *(const v8bf*)(qrow + 8  + klane);
    v8bf x2 = *(const v8bf*)(qrow + 32 + klane);
    v8bf x3 = *(const v8bf*)(qrow + 40 + klane);
#pragma unroll
    for (int i = 0; i < 8; ++i) {
      aq0[i] = x0[i]; aq0[8 + i] = x1[i];
      aq1[i] = x2[i]; aq1[8 + i] = x3[i];
    }
  }

  v8f o[4] = {v8f{}, v8f{}, v8f{}, v8f{}};

  for (int kt = 0; kt < kS; kt += 32) {
    // ---- V fragments (independent of LDS) issued first: their latency
    // ---- hides behind the score WMMAs and the softmax barrier phase.
    const int to = hg * 16;
    const bf16_t* vb = Vt + head_base + (size_t)lr * kS + kt + to;
    v16bf v0 = *(const v16bf*)(vb);
    v16bf v1 = *(const v16bf*)(vb + (size_t)16 * kS);
    v16bf v2 = *(const v16bf*)(vb + (size_t)32 * kS);
    v16bf v3 = *(const v16bf*)(vb + (size_t)48 * kS);

    // ---- scores: s = q @ k^T for 32 key tokens (two 16x16 accumulators) ----
    v8f s0{}, s1{};
    {
      const int co = hg * 16;
      const bf16_t* kr0 = Kb + head_base + (size_t)(kt + lr) * kDH;
      const bf16_t* kr1 = Kb + head_base + (size_t)(kt + 16 + lr) * kDH;
      __builtin_prefetch(kr0 + 32 * kDH, 0, 1);   // next k-tile
      v16bf b00 = *(const v16bf*)(kr0 + co);        // dk 0..31
      v16bf b01 = *(const v16bf*)(kr0 + 32 + co);   // dk 32..63
      v16bf b10 = *(const v16bf*)(kr1 + co);
      v16bf b11 = *(const v16bf*)(kr1 + 32 + co);
      s0 = WMMA_BF16(aq0, b00, s0);
      s0 = WMMA_BF16(aq1, b01, s0);
      s1 = WMMA_BF16(aq0, b10, s1);
      s1 = WMMA_BF16(aq1, b11, s1);
    }

    __syncthreads();  // previous tile's LDS reads complete
#pragma unroll
    for (int r = 0; r < 8; ++r) {
      const int m = r + hg * 8;
      sc[h][m][lr]      = s0[r] * scale;
      sc[h][m][16 + lr] = s1[r] * scale;
    }
    __syncthreads();  // all heads' scores visible

    // ---- cross-head softmax stats: one (q,k) pair per thread ----
    {
      const int q = threadIdx.x >> 5;
      const int k = threadIdx.x & 31;
      float mx = sc[0][q][k];
#pragma unroll
      for (int hh = 1; hh < kH; ++hh) mx = fmaxf(mx, sc[hh][q][k]);
      float sum = 0.0f;
#pragma unroll
      for (int hh = 0; hh < kH; ++hh) sum += __expf(sc[hh][q][k] - mx);
      mred[q][k] = mx;
      rinv[q][k] = 1.0f / sum;
    }
    __syncthreads();  // stats visible

    // ---- normalized weight A-fragment (16 q-rows x 32 k-tokens, bf16) ----
    v16bf aw;
    {
      const int klane = hg * 16;
#pragma unroll
      for (int i = 0; i < 8; ++i) {
        const int k1 = klane + i;
        const int k2 = klane + 8 + i;
        aw[i]     = (bf16_t)(__expf(sc[h][lr][k1] - mred[lr][k1]) * rinv[lr][k1]);
        aw[8 + i] = (bf16_t)(__expf(sc[h][lr][k2] - mred[lr][k2]) * rinv[lr][k2]);
      }
    }

    // ---- o += w @ v with the preloaded V fragments ----
    o[0] = WMMA_BF16(aw, v0, o[0]);
    o[1] = WMMA_BF16(aw, v1, o[1]);
    o[2] = WMMA_BF16(aw, v2, o[2]);
    o[3] = WMMA_BF16(aw, v3, o[3]);
  }

  // ---- epilogue: merged-head bf16 output [B*S, 1024], col = h*64 + d ----
#pragma unroll
  for (int j = 0; j < 4; ++j) {
    const int col = h * 64 + j * 16 + lr;
#pragma unroll
    for (int r = 0; r < 8; ++r) {
      const int m = r + hg * 8;
      const size_t row = (size_t)b * kS + q0 + m;
      Obf[row * kN + col] = (bf16_t)o[j][r];
    }
  }
}

// ---------------------------------------------------------------------------
// Host launcher
// ---------------------------------------------------------------------------
extern "C" void kernel_launch(void* const* d_in, const int* in_sizes, int n_in,
                              void* d_out, int out_size, void* d_ws, size_t ws_size,
                              hipStream_t stream) {
  const float* queries = (const float*)d_in[0];
  const float* keys    = (const float*)d_in[1];
  const float* values  = (const float*)d_in[2];
  const float* Wq = (const float*)d_in[3];
  const float* bq = (const float*)d_in[4];
  const float* Wk = (const float*)d_in[5];
  const float* bk = (const float*)d_in[6];
  const float* Wv = (const float*)d_in[7];
  const float* bv = (const float*)d_in[8];
  const float* Wo = (const float*)d_in[9];
  const float* bo = (const float*)d_in[10];
  float* outp = (float*)d_out;

  // Workspace partition (all offsets 256B aligned).
  char* ws = (char*)d_ws;
  size_t off = 0;
  auto take = [&](size_t bytes) -> char* {
    char* p = ws + off;
    off += (bytes + 255) & ~(size_t)255;
    return p;
  };
  const size_t w_bytes   = (size_t)kD * kN * sizeof(bf16_t);             // 2 MB
  const size_t qkv_bytes = (size_t)kB * kH * kS * kDH * sizeof(bf16_t);  // 8 MB
  bf16_t* WqT = (bf16_t*)take(w_bytes);
  bf16_t* WkT = (bf16_t*)take(w_bytes);
  bf16_t* WvT = (bf16_t*)take(w_bytes);
  bf16_t* WoT = (bf16_t*)take(w_bytes);
  bf16_t* Qb  = (bf16_t*)take(qkv_bytes);   // [B,H,S,64]
  bf16_t* Kb  = (bf16_t*)take(qkv_bytes);   // [B,H,S,64]
  bf16_t* Vt  = (bf16_t*)take(qkv_bytes);   // [B,H,64,S]
  bf16_t* Obf = (bf16_t*)take((size_t)kM * kN * sizeof(bf16_t));  // 8 MB
  (void)ws_size; (void)in_sizes; (void)n_in; (void)out_size;

  // 1) weight transpose + bf16 conversion
  const int wtot = kD * kN;
  wt_transpose_kernel<<<(wtot + 255) / 256, 256, 0, stream>>>(Wq, WqT);
  wt_transpose_kernel<<<(wtot + 255) / 256, 256, 0, stream>>>(Wk, WkT);
  wt_transpose_kernel<<<(wtot + 255) / 256, 256, 0, stream>>>(Wv, WvT);
  wt_transpose_kernel<<<(wtot + 255) / 256, 256, 0, stream>>>(Wo, WoT);

  // 2) Q/K/V projections (V stored transposed per head: ss=1, sd=S)
  dim3 gp(kM / 16, kN / 256);
  gemm_proj_kernel<<<gp, 128, 0, stream>>>(queries, WqT, bq, Qb, kDH, 1);
  gemm_proj_kernel<<<gp, 128, 0, stream>>>(keys,    WkT, bk, Kb, kDH, 1);
  gemm_proj_kernel<<<gp, 128, 0, stream>>>(values,  WvT, bv, Vt, 1, kS);

  // 3) attention, softmax over heads, streamed over k-tiles
  attn_kernel<<<dim3(kS / 16, kB), 512, 0, stream>>>(Qb, Kb, Vt, Obf);

  // 4) output projection (fp32 epilogue into d_out)
  gemm_out_kernel<<<dim3(kM / 16, kN / 256), 128, 0, stream>>>(Obf, WoT, bo, outp);
}
